// main_net_33895881900039
// MI455X (gfx1250) — matrix operand order repaired and checked by reading, not compiled, not run on hardware
//
#include <hip/hip_runtime.h>
#include <hip/hip_bf16.h>
#include <math.h>

// ---------------------------------------------------------------------------
// Model constants (from reference)
// ---------------------------------------------------------------------------
#define BATCH     4
#define NPTS      1023
#define MTOK      1024            // 1 cls + 1023 tokens
#define DMODEL    256
#define NHEADS    8
#define DHEAD     32
#define NLAYERS   6
#define FFNH      1024
#define OUTDIM    15              // 3^2 + 2*3
#define TOTROWS   (BATCH * MTOK)  // 4096

typedef __bf16 bf16_t;
typedef __attribute__((ext_vector_type(16))) __bf16 v16bf;
typedef __attribute__((ext_vector_type(8)))  __bf16 v8bf;
typedef __attribute__((ext_vector_type(8)))  float  v8f;

__device__ __forceinline__ float gelu_f(float x) {
    return 0.5f * x * (1.0f + erff(x * 0.70710678118654752440f));
}

// ---------------------------------------------------------------------------
// Embedding: tok = t2(gelu(t1([rm,xyz]))) + p2(gelu(p1(xyz))); x = LN(tok)
// Block = 256 threads, one block per (b, token). Token 0 = cls (no LN).
// Also writes xyz (with pole at index 0) used for relative-position bias.
// ---------------------------------------------------------------------------
__global__ __launch_bounds__(256) void embed_kernel(
    const float* __restrict__ data, const float* __restrict__ cls,
    const float* __restrict__ t1w, const float* __restrict__ t1b,
    const float* __restrict__ t2w, const float* __restrict__ t2b,
    const float* __restrict__ p1w, const float* __restrict__ p1b,
    const float* __restrict__ p2w, const float* __restrict__ p2b,
    const float* __restrict__ lng, const float* __restrict__ lnb,
    float* __restrict__ x, float* __restrict__ xyz)
{
    const int tok = blockIdx.x;          // b*1024 + i
    const int b   = tok >> 10;
    const int i   = tok & 1023;
    const int j   = threadIdx.x;

    __shared__ float sh[DMODEL];
    __shared__ float red[DMODEL];
    __shared__ float base[4];
    __shared__ float pp[32];

    if (i == 0) {                        // cls token, pole position
        x[(size_t)tok * DMODEL + j] = cls[j];
        if (j < 3) xyz[tok * 3 + j] = (j == 2) ? 1.0f : 0.0f;
        return;
    }
    const int src = i - 1;
    if (j == 0) {
        float th = data[(b * 3 + 0) * NPTS + src];
        float ph = data[(b * 3 + 1) * NPTS + src];
        float rm = data[(b * 3 + 2) * NPTS + src];
        float st = sinf(th);
        float x0 = st * cosf(ph), x1 = st * sinf(ph), x2 = cosf(th);
        base[0] = rm; base[1] = x0; base[2] = x1; base[3] = x2;
        xyz[tok * 3 + 0] = x0; xyz[tok * 3 + 1] = x1; xyz[tok * 3 + 2] = x2;
    }
    __syncthreads();

    float h1 = t1b[j];
    for (int c = 0; c < 4; ++c) h1 += base[c] * t1w[c * DMODEL + j];
    sh[j] = gelu_f(h1);
    if (j < 32) {
        float v = p1b[j];
        for (int c = 0; c < 3; ++c) v += base[1 + c] * p1w[c * 32 + j];
        pp[j] = gelu_f(v);
    }
    __syncthreads();

    float t = t2b[j];
    for (int k = 0; k < DMODEL; ++k) t += sh[k] * t2w[k * DMODEL + j];
    float t2 = p2b[j];
    for (int k = 0; k < 32; ++k) t2 += pp[k] * p2w[k * DMODEL + j];
    t += t2;

    // LayerNorm over 256
    red[j] = t; __syncthreads();
    for (int s = 128; s > 0; s >>= 1) { if (j < s) red[j] += red[j + s]; __syncthreads(); }
    float mu = red[0] * (1.0f / DMODEL); __syncthreads();
    float d = t - mu;
    red[j] = d * d; __syncthreads();
    for (int s = 128; s > 0; s >>= 1) { if (j < s) red[j] += red[j + s]; __syncthreads(); }
    float var = red[0] * (1.0f / DMODEL);
    x[(size_t)tok * DMODEL + j] = d * rsqrtf(var + 1e-5f) * lng[j] + lnb[j];
}

// ---------------------------------------------------------------------------
// LayerNorm: f32 (rows x 256) -> bf16, one block per row
// ---------------------------------------------------------------------------
__global__ __launch_bounds__(256) void ln_kernel(
    const float* __restrict__ x, const float* __restrict__ g,
    const float* __restrict__ bb, bf16_t* __restrict__ out)
{
    const int row = blockIdx.x;
    const int j = threadIdx.x;
    __shared__ float red[DMODEL];
    float v = x[(size_t)row * DMODEL + j];
    red[j] = v; __syncthreads();
    for (int s = 128; s > 0; s >>= 1) { if (j < s) red[j] += red[j + s]; __syncthreads(); }
    float mu = red[0] * (1.0f / DMODEL); __syncthreads();
    float d = v - mu;
    red[j] = d * d; __syncthreads();
    for (int s = 128; s > 0; s >>= 1) { if (j < s) red[j] += red[j + s]; __syncthreads(); }
    float var = red[0] * (1.0f / DMODEL);
    out[(size_t)row * DMODEL + j] = (bf16_t)(d * rsqrtf(var + 1e-5f) * g[j] + bb[j]);
}

// ---------------------------------------------------------------------------
// Weight prep: (K,N) f32 row-major -> (N,K) bf16  (K contiguous for WMMA B)
// ---------------------------------------------------------------------------
__global__ void wprep_kernel(const float* __restrict__ w, bf16_t* __restrict__ wt,
                             int K, int N)
{
    int idx = blockIdx.x * 256 + threadIdx.x;
    if (idx >= K * N) return;
    int k = idx / N, n = idx % N;
    wt[(size_t)n * K + k] = (bf16_t)w[idx];
}

// ---------------------------------------------------------------------------
// Relative-position bias precompute: bias[b,h,n,m] = b2(relu(b1(xyz_n-xyz_m)))
// One thread per (b,n,m); hidden(32) shared across all 8 heads.
// Layout [b][h][n][m] so attention reads are lane-contiguous in m.
// ---------------------------------------------------------------------------
__global__ __launch_bounds__(256) void bias_kernel(
    const float* __restrict__ xyz,
    const float* __restrict__ w1, const float* __restrict__ b1,
    const float* __restrict__ w2, const float* __restrict__ b2,
    bf16_t* __restrict__ out)
{
    size_t idx = (size_t)blockIdx.x * 256 + threadIdx.x;  // b*M*M + n*M + m
    int m = (int)(idx & 1023);
    size_t t = idx >> 10;
    int n = (int)(t & 1023);
    int b = (int)(t >> 10);
    const float* pn = xyz + (size_t)(b * MTOK + n) * 3;
    const float* pm = xyz + (size_t)(b * MTOK + m) * 3;
    float r0 = pn[0] - pm[0], r1 = pn[1] - pm[1], r2 = pn[2] - pm[2];
    float hd[32];
    #pragma unroll
    for (int jj = 0; jj < 32; ++jj) {
        float h = b1[jj] + r0 * w1[jj] + r1 * w1[32 + jj] + r2 * w1[64 + jj];
        hd[jj] = fmaxf(h, 0.0f);
    }
    #pragma unroll
    for (int h = 0; h < NHEADS; ++h) {
        float a = b2[h];
        #pragma unroll
        for (int jj = 0; jj < 32; ++jj) a += hd[jj] * w2[jj * NHEADS + h];
        out[(((size_t)(b * NHEADS + h) * MTOK + n) * MTOK) + m] = (bf16_t)a;
    }
}

// ---------------------------------------------------------------------------
// bf16 WMMA GEMM:  out = act(A[M,K] * Wt[N,K]^T + bias) (+resid)
// 256 threads = 8 waves/block; each wave computes a 16x64 C strip:
// one A fragment reused across 4 B fragments (4 v8f accumulators).
// Fragment layouts used here (wave32; A and B share one contiguous k order):
//   A: M=lane%16, e -> K=(lane/16)*16+e               (2x contiguous v8bf)
//   B: N=lane%16, e -> K=(lane/16)*16+e               (1x contiguous v16bf)
//   C: vgpr r     -> M=r+8*(lane/16), N=lane%16
// Optional outBT writes the per-head transposed V layout vt[b][h][d][m].
// ---------------------------------------------------------------------------
__global__ __launch_bounds__(256) void gemm_bf16_kernel(
    const bf16_t* __restrict__ A, const bf16_t* __restrict__ Wt,
    const float* __restrict__ bias, const float* __restrict__ resid,
    float* __restrict__ outF, bf16_t* __restrict__ outB,
    bf16_t* __restrict__ outBT,
    int M, int N, int K, int doGelu)
{
    const int wave = threadIdx.x >> 5;
    const int lane = threadIdx.x & 31;
    const int tile = blockIdx.x * 8 + wave;     // 16x64 strips
    const int tilesN = N >> 6;
    const int tm = tile / tilesN;
    const int tn = tile - tm * tilesN;
    const int ln16 = lane & 15;
    const int hk   = lane >> 4;                 // 0 or 1

    const bf16_t* ap = A + (size_t)(tm * 16 + ln16) * K + hk * 16;
    const bf16_t* bp0 = Wt + (size_t)(tn * 64 +  0 + ln16) * K + hk * 16;
    const bf16_t* bp1 = Wt + (size_t)(tn * 64 + 16 + ln16) * K + hk * 16;
    const bf16_t* bp2 = Wt + (size_t)(tn * 64 + 32 + ln16) * K + hk * 16;
    const bf16_t* bp3 = Wt + (size_t)(tn * 64 + 48 + ln16) * K + hk * 16;

    v8f c0 = {}, c1 = {}, c2 = {}, c3 = {};
    for (int k = 0; k < K; k += 32) {
        v8bf alo = *(const v8bf*)(ap + k);
        v8bf ahi = *(const v8bf*)(ap + k + 8);
        v16bf a = __builtin_shufflevector(alo, ahi,
                    0,1,2,3,4,5,6,7,8,9,10,11,12,13,14,15);
        v16bf b0 = *(const v16bf*)(bp0 + k);
        v16bf b1 = *(const v16bf*)(bp1 + k);
        v16bf b2 = *(const v16bf*)(bp2 + k);
        v16bf b3 = *(const v16bf*)(bp3 + k);
        c0 = __builtin_amdgcn_wmma_f32_16x16x32_bf16(false, a, false, b0, (short)0, c0, false, false);
        c1 = __builtin_amdgcn_wmma_f32_16x16x32_bf16(false, a, false, b1, (short)0, c1, false, false);
        c2 = __builtin_amdgcn_wmma_f32_16x16x32_bf16(false, a, false, b2, (short)0, c2, false, false);
        c3 = __builtin_amdgcn_wmma_f32_16x16x32_bf16(false, a, false, b3, (short)0, c3, false, false);
    }

    v8f cs[4] = {c0, c1, c2, c3};
    #pragma unroll
    for (int s = 0; s < 4; ++s) {
        const int nOut = tn * 64 + s * 16 + ln16;
        #pragma unroll
        for (int r = 0; r < 8; ++r) {
            int mOut = tm * 16 + r + hk * 8;
            float v = cs[s][r] + bias[nOut];
            if (doGelu) v = gelu_f(v);
            size_t o = (size_t)mOut * N + nOut;
            if (resid) v += resid[o];
            if (outF) outF[o] = v;
            if (outB) outB[o] = (bf16_t)v;
            if (outBT) {
                // vt[b][h][d][m]: row mOut = b*1024 + t, col nOut = h*32 + d
                int bb2 = mOut >> 10, tt = mOut & 1023;
                int hh = nOut >> 5,  dd = nOut & 31;
                outBT[(((size_t)(bb2 * NHEADS + hh) * DHEAD + dd) * MTOK) + tt] = (bf16_t)v;
            }
        }
    }
}

// ---------------------------------------------------------------------------
// Attention: one wave per (b, head, 16-query tile).
// scores(16x1024) via WMMA QK^T + bias, kept in 64KB LDS; shfl softmax;
// AV via WMMA on transposed V (contiguous B fragments), 1/rowsum folded
// into the A (probability) fragment assembled from 32-byte LDS loads.
// ---------------------------------------------------------------------------
__global__ __launch_bounds__(32) void attn_kernel(
    const bf16_t* __restrict__ q, const bf16_t* __restrict__ k,
    const bf16_t* __restrict__ vt,            // vt[b][h][d][m]
    const bf16_t* __restrict__ biasbuf,       // [b][h][n][m] or null
    const float* __restrict__ xyz,
    const float* __restrict__ b1w, const float* __restrict__ b1b,
    const float* __restrict__ b2w, const float* __restrict__ b2b,
    bf16_t* __restrict__ out)
{
    __shared__ __align__(32) float sc[16 * MTOK];   // 64 KB of 320 KB WGP LDS

    const int bid = blockIdx.x;          // b*8*64 + h*64 + tile
    const int tq  = bid & 63;
    const int h   = (bid >> 6) & 7;
    const int b   = bid >> 9;
    const int n0  = tq * 16;
    const int lane = threadIdx.x;
    const int ln16 = lane & 15;
    const int hk   = lane >> 4;
    const float scale = 0.17677669529663687f;   // 1/sqrt(32)

    // Q fragment (A matrix, 16x32)
    const bf16_t* qp = q + (size_t)(b * MTOK + n0 + ln16) * DMODEL + h * DHEAD + hk * 16;
    v8bf qlo = *(const v8bf*)qp;
    v8bf qhi = *(const v8bf*)(qp + 8);
    v16bf qa = __builtin_shufflevector(qlo, qhi,
                 0,1,2,3,4,5,6,7,8,9,10,11,12,13,14,15);

    const bf16_t* biasrow = biasbuf
        ? biasbuf + ((size_t)(b * NHEADS + h) * MTOK) * MTOK : nullptr;

    for (int m0 = 0; m0 < MTOK; m0 += 16) {
        const bf16_t* kp = k + (size_t)(b * MTOK + m0 + ln16) * DMODEL + h * DHEAD + hk * 16;
        v16bf kb = *(const v16bf*)kp;
        v8f c = {};
        c = __builtin_amdgcn_wmma_f32_16x16x32_bf16(
                false, qa, false, kb, (short)0, c, false, false);
        #pragma unroll
        for (int r = 0; r < 8; ++r) {
            int nl = r + hk * 8;
            int n  = n0 + nl;
            int m  = m0 + ln16;
            float s = c[r] * scale;
            if (biasrow) {
                s += (float)biasrow[(size_t)n * MTOK + m];
            } else {
                const float* pn = xyz + (size_t)(b * MTOK + n) * 3;
                const float* pm = xyz + (size_t)(b * MTOK + m) * 3;
                float r0 = pn[0] - pm[0], r1 = pn[1] - pm[1], r2 = pn[2] - pm[2];
                float acc = b2b[h];
                for (int jj = 0; jj < 32; ++jj) {
                    float t = b1b[jj] + r0 * b1w[jj] + r1 * b1w[32 + jj] + r2 * b1w[64 + jj];
                    acc += fmaxf(t, 0.0f) * b2w[jj * NHEADS + h];
                }
                s += acc;
            }
            sc[nl * MTOK + m] = s;
        }
    }
    __syncthreads();

    // softmax (rows of 1024), keep 1/sum for this lane's A-row (ln16)
    float myInv = 1.0f;
    for (int r = 0; r < 16; ++r) {
        float mx = -1e30f;
        for (int m = lane; m < MTOK; m += 32) mx = fmaxf(mx, sc[r * MTOK + m]);
        for (int o = 16; o > 0; o >>= 1) mx = fmaxf(mx, __shfl_xor(mx, o, 32));
        float sum = 0.0f;
        for (int m = lane; m < MTOK; m += 32) {
            float e = __expf(sc[r * MTOK + m] - mx);
            sc[r * MTOK + m] = e;
            sum += e;
        }
        for (int o = 16; o > 0; o >>= 1) sum += __shfl_xor(sum, o, 32);
        if (r == ln16) myInv = 1.0f / sum;
    }
    __syncthreads();

    // out = P * V   (K dim = 1024 keys, N dim = 32 head channels -> 2 tiles)
    const float* prow = &sc[ln16 * MTOK];
    const bf16_t* vrow0 = vt + ((size_t)(b * NHEADS + h) * DHEAD + ln16)      * MTOK + hk * 16;
    const bf16_t* vrow1 = vt + ((size_t)(b * NHEADS + h) * DHEAD + 16 + ln16) * MTOK + hk * 16;
    v8f acc0 = {}, acc1 = {};
    for (int k0 = 0; k0 < MTOK; k0 += 32) {
        v8f plo = *(const v8f*)(prow + k0 + hk * 16);
        v8f phi = *(const v8f*)(prow + k0 + hk * 16 + 8);
        v16bf a;
        #pragma unroll
        for (int e = 0; e < 8; ++e) {
            a[e]     = (bf16_t)(plo[e] * myInv);
            a[e + 8] = (bf16_t)(phi[e] * myInv);
        }
        v16bf b0 = *(const v16bf*)(vrow0 + k0);
        v16bf b1 = *(const v16bf*)(vrow1 + k0);
        acc0 = __builtin_amdgcn_wmma_f32_16x16x32_bf16(
                   false, a, false, b0, (short)0, acc0, false, false);
        acc1 = __builtin_amdgcn_wmma_f32_16x16x32_bf16(
                   false, a, false, b1, (short)0, acc1, false, false);
    }
    #pragma unroll
    for (int r = 0; r < 8; ++r) {
        int nl = r + hk * 8;
        bf16_t* op = out + (size_t)(b * MTOK + n0 + nl) * DMODEL + h * DHEAD;
        op[ln16]      = (bf16_t)acc0[r];
        op[16 + ln16] = (bf16_t)acc1[r];
    }
}

// ---------------------------------------------------------------------------
// Head: pooled = LN(x[:,0,:]); out = r2(gelu(r1(pooled))). One block / batch.
// ---------------------------------------------------------------------------
__global__ __launch_bounds__(256) void head_kernel(
    const float* __restrict__ x,
    const float* __restrict__ lng, const float* __restrict__ lnb,
    const float* __restrict__ r1w, const float* __restrict__ r1b,
    const float* __restrict__ r2w, const float* __restrict__ r2b,
    float* __restrict__ out)
{
    const int b = blockIdx.x;
    const int j = threadIdx.x;
    __shared__ float p[DMODEL];
    __shared__ float red[DMODEL];
    __shared__ float g[128];
    float v = x[(size_t)(b * MTOK) * DMODEL + j];
    red[j] = v; __syncthreads();
    for (int s = 128; s > 0; s >>= 1) { if (j < s) red[j] += red[j + s]; __syncthreads(); }
    float mu = red[0] * (1.0f / DMODEL); __syncthreads();
    float d = v - mu;
    red[j] = d * d; __syncthreads();
    for (int s = 128; s > 0; s >>= 1) { if (j < s) red[j] += red[j + s]; __syncthreads(); }
    float var = red[0] * (1.0f / DMODEL);
    p[j] = d * rsqrtf(var + 1e-5f) * lng[j] + lnb[j];
    __syncthreads();
    if (j < 128) {
        float a = r1b[j];
        for (int kk = 0; kk < DMODEL; ++kk) a += p[kk] * r1w[kk * 128 + j];
        g[j] = gelu_f(a);
    }
    __syncthreads();
    if (j < OUTDIM) {
        float a = r2b[j];
        for (int kk = 0; kk < 128; ++kk) a += g[kk] * r2w[kk * OUTDIM + j];
        out[b * OUTDIM + j] = a;
    }
}

// ---------------------------------------------------------------------------
// Host orchestration
// ---------------------------------------------------------------------------
extern "C" void kernel_launch(void* const* d_in, const int* in_sizes, int n_in,
                              void* d_out, int out_size, void* d_ws, size_t ws_size,
                              hipStream_t stream)
{
    (void)in_sizes; (void)n_in; (void)out_size;
    auto F = [&](int i) { return (const float*)d_in[i]; };

    // Input leaf order (JAX pytree: sorted dict keys):
    // 0: data, 1: cls,
    // per layer (base 2 + 20*l):
    //   +0 b1.b +1 b1.w +2 b2.b +3 b2.w +4 f1.b +5 f1.w +6 f2.b +7 f2.w
    //   +8 k.b  +9 k.w  +10 ln1_b +11 ln1_g +12 ln2_b +13 ln2_g
    //   +14 o.b +15 o.w +16 q.b +17 q.w +18 v.b +19 v.w
    // reg (122..127): ln_b, ln_g, r1.b, r1.w, r2.b, r2.w
    // token (128..137): ln_b, ln_g, p1.b, p1.w, p2.b, p2.w, t1.b, t1.w, t2.b, t2.w
    const int RB = 2 + NLAYERS * 20;   // 122
    const int TB = RB + 6;             // 128

    // Workspace layout
    char* ws = (char*)d_ws;
    size_t off = 0;
    auto alloc = [&](size_t bytes) {
        size_t o = off;
        off = (off + bytes + 255) & ~(size_t)255;
        return o;
    };
    float*  xf   = (float*)(ws + alloc((size_t)TOTROWS * DMODEL * 4));
    float*  xyz  = (float*)(ws + alloc((size_t)TOTROWS * 3 * 4));
    bf16_t* hbf  = (bf16_t*)(ws + alloc((size_t)TOTROWS * DMODEL * 2));
    bf16_t* qbf  = (bf16_t*)(ws + alloc((size_t)TOTROWS * DMODEL * 2));
    bf16_t* kbf  = (bf16_t*)(ws + alloc((size_t)TOTROWS * DMODEL * 2));
    bf16_t* vtbf = (bf16_t*)(ws + alloc((size_t)TOTROWS * DMODEL * 2));  // vt[b][h][d][m]
    bf16_t* aobf = (bf16_t*)(ws + alloc((size_t)TOTROWS * DMODEL * 2));
    bf16_t* ffbf = (bf16_t*)(ws + alloc((size_t)TOTROWS * FFNH * 2));
    bf16_t* wq   = (bf16_t*)(ws + alloc((size_t)DMODEL * DMODEL * 2));
    bf16_t* wk   = (bf16_t*)(ws + alloc((size_t)DMODEL * DMODEL * 2));
    bf16_t* wv   = (bf16_t*)(ws + alloc((size_t)DMODEL * DMODEL * 2));
    bf16_t* wo   = (bf16_t*)(ws + alloc((size_t)DMODEL * DMODEL * 2));
    bf16_t* wf1  = (bf16_t*)(ws + alloc((size_t)DMODEL * FFNH * 2));
    bf16_t* wf2  = (bf16_t*)(ws + alloc((size_t)FFNH * DMODEL * 2));

    // Optional precomputed attention bias (67 MB) — fall back to inline VALU
    bf16_t* biasbuf = nullptr;
    const size_t biasBytes = (size_t)BATCH * NHEADS * MTOK * MTOK * 2;
    if (ws_size >= off + biasBytes)
        biasbuf = (bf16_t*)(ws + alloc(biasBytes));

    // ---- Embedding ----
    embed_kernel<<<TOTROWS, 256, 0, stream>>>(
        F(0), F(1),
        F(TB + 7), F(TB + 6),   // t1.w, t1.b
        F(TB + 9), F(TB + 8),   // t2.w, t2.b
        F(TB + 3), F(TB + 2),   // p1.w, p1.b
        F(TB + 5), F(TB + 4),   // p2.w, p2.b
        F(TB + 1), F(TB + 0),   // ln_g, ln_b
        xf, xyz);

    // Grid sizes: each block = 8 waves, each wave = one 16x64 C strip
    const int blksD = (TOTROWS / 16) * (DMODEL / 64) / 8;   // 128
    const int blksF = (TOTROWS / 16) * (FFNH / 64) / 8;     // 512

    // ---- Transformer layers ----
    for (int l = 0; l < NLAYERS; ++l) {
        const int L = 2 + l * 20;
        const float *b1b = F(L + 0),  *b1w = F(L + 1);
        const float *b2b = F(L + 2),  *b2w = F(L + 3);
        const float *f1b = F(L + 4),  *f1w = F(L + 5);
        const float *f2b = F(L + 6),  *f2w = F(L + 7);
        const float *kb  = F(L + 8),  *kw  = F(L + 9);
        const float *ln1b = F(L + 10), *ln1g = F(L + 11);
        const float *ln2b = F(L + 12), *ln2g = F(L + 13);
        const float *ob  = F(L + 14), *ow  = F(L + 15);
        const float *qb  = F(L + 16), *qw  = F(L + 17);
        const float *vb  = F(L + 18), *vw  = F(L + 19);

        // weight prep (transpose + bf16)
        wprep_kernel<<<(DMODEL * DMODEL + 255) / 256, 256, 0, stream>>>(qw, wq, DMODEL, DMODEL);
        wprep_kernel<<<(DMODEL * DMODEL + 255) / 256, 256, 0, stream>>>(kw, wk, DMODEL, DMODEL);
        wprep_kernel<<<(DMODEL * DMODEL + 255) / 256, 256, 0, stream>>>(vw, wv, DMODEL, DMODEL);
        wprep_kernel<<<(DMODEL * DMODEL + 255) / 256, 256, 0, stream>>>(ow, wo, DMODEL, DMODEL);
        wprep_kernel<<<(DMODEL * FFNH + 255) / 256, 256, 0, stream>>>(f1w, wf1, DMODEL, FFNH);
        wprep_kernel<<<(FFNH * DMODEL + 255) / 256, 256, 0, stream>>>(f2w, wf2, FFNH, DMODEL);

        if (biasbuf)
            bias_kernel<<<(BATCH * MTOK * MTOK) / 256, 256, 0, stream>>>(
                xyz, b1w, b1b, b2w, b2b, biasbuf);

        // h = LN1(x)
        ln_kernel<<<TOTROWS, 256, 0, stream>>>(xf, ln1g, ln1b, hbf);

        // q/k/v projections (V written directly in transposed per-head layout)
        gemm_bf16_kernel<<<blksD, 256, 0, stream>>>(
            hbf, wq, qb, nullptr, nullptr, qbf, nullptr, TOTROWS, DMODEL, DMODEL, 0);
        gemm_bf16_kernel<<<blksD, 256, 0, stream>>>(
            hbf, wk, kb, nullptr, nullptr, kbf, nullptr, TOTROWS, DMODEL, DMODEL, 0);
        gemm_bf16_kernel<<<blksD, 256, 0, stream>>>(
            hbf, wv, vb, nullptr, nullptr, nullptr, vtbf, TOTROWS, DMODEL, DMODEL, 0);

        // attention
        attn_kernel<<<BATCH * NHEADS * (MTOK / 16), 32, 0, stream>>>(
            qbf, kbf, vtbf, biasbuf, xyz, b1w, b1b, b2w, b2b, aobf);

        // x = x + O(attn_out)
        gemm_bf16_kernel<<<blksD, 256, 0, stream>>>(
            aobf, wo, ob, xf, xf, nullptr, nullptr, TOTROWS, DMODEL, DMODEL, 0);

        // h2 = LN2(x)
        ln_kernel<<<TOTROWS, 256, 0, stream>>>(xf, ln2g, ln2b, hbf);

        // FFN
        gemm_bf16_kernel<<<blksF, 256, 0, stream>>>(
            hbf, wf1, f1b, nullptr, nullptr, ffbf, nullptr, TOTROWS, FFNH, DMODEL, 1);
        gemm_bf16_kernel<<<blksD, 256, 0, stream>>>(
            ffbf, wf2, f2b, xf, xf, nullptr, nullptr, TOTROWS, DMODEL, FFNH, 0);
    }

    // ---- Regression head ----
    head_kernel<<<BATCH, 256, 0, stream>>>(
        xf, F(RB + 1), F(RB + 0), F(RB + 3), F(RB + 2), F(RB + 5), F(RB + 4),
        (float*)d_out);
}
